// PointReg_6502580486422
// MI455X (gfx1250) — matrix operand order repaired and checked by reading, not compiled, not run on hardware
//
#include <hip/hip_runtime.h>
#include <hip/hip_bf16.h>
#include <float.h>
#include <math.h>

// ---------------------------------------------------------------------------
// PointNet++-MSG dual-branch forward for gfx1250 (wave32, WMMA).
// GEMMs: v_wmma_f32_16x16x32_f16, guard-free fragment loads:
//   A: padded f16 activations, two 16B loads/lane (ISA 16-bit A layout)
//   B: pre-packed f16 weights, one 32B coalesced load/lane
// ---------------------------------------------------------------------------

#define TPB 256

typedef __attribute__((ext_vector_type(16))) _Float16 v16h;
typedef __attribute__((ext_vector_type(8)))  _Float16 v8h;
typedef __attribute__((ext_vector_type(8)))  float    v8f;

// (B,C,N) -> (B,N,C)
__global__ void k_transpose(const float* __restrict__ src, float* __restrict__ dst,
                            int B, int C, int N) {
  int tid = blockIdx.x * blockDim.x + threadIdx.x;
  int total = B * C * N;
  if (tid >= total) return;
  int c = tid % C;
  int n = (tid / C) % N;
  int b = tid / (C * N);
  dst[(b * N + n) * C + c] = src[(b * C + c) * N + n];
}

// Farthest point sampling; one block per batch; dist lives in LDS. N <= 2048.
__global__ void k_fps(const float* __restrict__ pts, int* __restrict__ out_idx,
                      int N, int C, int npoint) {
  __shared__ float dist[2048];
  __shared__ float rv[TPB];
  __shared__ int   ri[TPB];
  int b = blockIdx.x;
  int t = threadIdx.x;
  for (int j = t; j < N; j += TPB) dist[j] = 1e10f;
  __syncthreads();
  int far = 0;
  for (int i = 0; i < npoint; i++) {
    if (t == 0) out_idx[b * npoint + i] = far;
    float cx0 = pts[(b * N + far) * C + 0];
    float cx1 = pts[(b * N + far) * C + 1];
    float cx2 = (C > 2) ? pts[(b * N + far) * C + 2] : 0.f;
    float best = -1.f;
    int   bi = 0;
    for (int j = t; j < N; j += TPB) {
      const float* p = &pts[(b * N + j) * C];
      float d0 = p[0] - cx0, d1 = p[1] - cx1;
      float d = d0 * d0 + d1 * d1;
      if (C > 2) { float d2 = p[2] - cx2; d += d2 * d2; }
      float nd = fminf(dist[j], d);
      dist[j] = nd;
      if (nd > best) { best = nd; bi = j; }   // strict > keeps first occurrence
    }
    rv[t] = best; ri[t] = bi;
    __syncthreads();
    for (int s = TPB / 2; s > 0; s >>= 1) {
      if (t < s) {
        if (rv[t + s] > rv[t] || (rv[t + s] == rv[t] && ri[t + s] < ri[t])) {
          rv[t] = rv[t + s]; ri[t] = ri[t + s];
        }
      }
      __syncthreads();
    }
    far = ri[0];
    __syncthreads();
  }
}

// out[b,m,c] = pts[b, idx[b,m], c]
__global__ void k_gather(const float* __restrict__ pts, const int* __restrict__ idx,
                         float* __restrict__ out, int B, int N, int C, int M) {
  int tid = blockIdx.x * blockDim.x + threadIdx.x;
  int total = B * M * C;
  if (tid >= total) return;
  int c = tid % C;
  int m = (tid / C) % M;
  int b = tid / (C * M);
  out[tid] = pts[(b * N + idx[b * M + m]) * C + c];
}

// Ball query: first K in-range indices ascending; pad with first hit.
__global__ void k_query_ball(const float* __restrict__ pts, const float* __restrict__ ctr,
                             int* __restrict__ gi, int B, int N, int C, int M, int K,
                             float r2) {
  int tid = blockIdx.x * blockDim.x + threadIdx.x;
  if (tid >= B * M) return;
  int b = tid / M, m = tid % M;
  float c0 = ctr[(b * M + m) * C + 0];
  float c1 = ctr[(b * M + m) * C + 1];
  float c2 = (C > 2) ? ctr[(b * M + m) * C + 2] : 0.f;
  int* g = &gi[(long)(b * M + m) * K];
  int cnt = 0, first = 0;
  for (int j = 0; j < N && cnt < K; j++) {
    const float* p = &pts[(b * N + j) * C];
    float d0 = p[0] - c0, d1 = p[1] - c1;
    float d = d0 * d0 + d1 * d1;
    if (C > 2) { float d2 = p[2] - c2; d += d2 * d2; }
    if (d <= r2) { if (cnt == 0) first = j; g[cnt++] = j; }
  }
  for (int k = cnt; k < K; k++) g[k] = first;
}

// Build grouped activation (f16, zero-padded to rowsPad x Ctpad):
// value = concat(feats[b,gi,:], pts[b,gi,:] - new_xyz[b,m,:])
__global__ void k_group_f16(const float* __restrict__ pts, const float* __restrict__ feats,
                            const float* __restrict__ nx, const int* __restrict__ gi,
                            _Float16* __restrict__ out, int B, int N, int Cx, int Cf,
                            int M, int K, int Ctpad, long rowsPad) {
  long tid = (long)blockIdx.x * blockDim.x + threadIdx.x;
  long total = rowsPad * Ctpad;
  if (tid >= total) return;
  int  c = (int)(tid % Ctpad);
  long r = tid / Ctpad;
  int  Ct = Cf + Cx;
  long rows = (long)B * M * K;
  float v = 0.f;
  if (r < rows && c < Ct) {
    int k = (int)(r % K);
    long r2 = r / K;
    int m = (int)(r2 % M);
    int b = (int)(r2 / M);
    int j = gi[((long)b * M + m) * K + k];
    if (c < Cf) v = feats[((long)b * N + j) * Cf + c];
    else { int cc = c - Cf; v = pts[((long)b * N + j) * Cx + cc] - nx[((long)b * M + m) * Cx + cc]; }
  }
  out[tid] = (_Float16)v;
}

// Concat two f32 row-major sources into padded f16 activation (src2 may be null).
__global__ void k_concat2_f16(const float* __restrict__ s1, const float* __restrict__ s2,
                              _Float16* __restrict__ out, long rows, int C1, int C2,
                              int Cpad, long rowsPad) {
  long tid = (long)blockIdx.x * blockDim.x + threadIdx.x;
  long total = rowsPad * Cpad;
  if (tid >= total) return;
  int  c = (int)(tid % Cpad);
  long r = tid / Cpad;
  float v = 0.f;
  if (r < rows) {
    if (c < C1)            v = s1[r * C1 + c];
    else if (c < C1 + C2)  v = s2[r * C2 + (c - C1)];
  }
  out[tid] = (_Float16)v;
}

// Pack W[Kd x Nc] (f32) into B-fragment layout Wp[tN][kblk][lane][16] (f16).
// element h holds K = kblk*32 + 16*(h>>3) + 8*(lane>>4) + (h&7), col = tN*16 + (lane&15).
__global__ void k_pack_w(const float* __restrict__ W, _Float16* __restrict__ Wp,
                         int Kd, int Nc, int Kpad) {
  int KB = Kpad >> 5;
  long tid = (long)blockIdx.x * blockDim.x + threadIdx.x;
  long total = (long)(Nc >> 4) * KB * 32 * 16;
  if (tid >= total) return;
  int  h    = (int)(tid & 15);
  int  lane = (int)((tid >> 4) & 31);
  long r    = tid >> 9;
  int  kblk = (int)(r % KB);
  int  tN   = (int)(r / KB);
  int  k    = kblk * 32 + ((h >> 3) << 4) + ((lane >> 4) << 3) + (h & 7);
  int  col  = tN * 16 + (lane & 15);
  Wp[tid] = (k < Kd) ? (_Float16)W[(long)k * Nc + col] : (_Float16)0.f;
}

// H[Mrows,Nc] = A[*,Kpad](f16, padded) * Wpacked + bias. One wave = 16x16 tile.
__global__ void k_gemm_wmma(const _Float16* __restrict__ A, const _Float16* __restrict__ Wp,
                            const float* __restrict__ bias, float* __restrict__ H,
                            int Mrows, int Kpad, int Nc, int tiles) {
  int wave = blockIdx.x * (blockDim.x >> 5) + (threadIdx.x >> 5);
  if (wave >= tiles) return;                   // wave-uniform: EXEC stays all-1s
  int lane = threadIdx.x & 31;
  int nt = Nc >> 4;
  int tM = wave / nt, tN = wave % nt;
  int hi = lane >> 4;                          // 0: lanes 0-15, 1: lanes 16-31
  const _Float16* Arow = A + (long)(tM * 16 + (lane & 15)) * Kpad;
  int KB = Kpad >> 5;
  const _Float16* Bp = Wp + ((long)tN * KB * 32 + lane) * 16;
  v8f acc = {};
  for (int kb = 0; kb < KB; kb++) {
    __builtin_prefetch(Bp + (long)(kb + 1) * 512, 0, 1);
    // A fragment (16-bit A 16x32 layout): two contiguous 8-half chunks.
    v8h a0 = *(const v8h*)(Arow + (kb << 5) + (hi << 3));
    v8h a1 = *(const v8h*)(Arow + (kb << 5) + 16 + (hi << 3));
    v16h a = __builtin_shufflevector(a0, a1, 0, 1, 2, 3, 4, 5, 6, 7,
                                     8, 9, 10, 11, 12, 13, 14, 15);
    // B fragment: 16 contiguous halves per lane, coalesced across wave.
    v16h b = *(const v16h*)(Bp + (long)kb * 512);
    acc = __builtin_amdgcn_wmma_f32_16x16x32_f16(false, a, false, b,
                                                 (short)0, acc, false, false);
  }
  int colB = tN * 16 + (lane & 15);
  float bv = bias[colB];
  int rbase = tM * 16 + (hi << 3);             // C layout: lanes0-15 M0-7, 16-31 M8-15
#pragma unroll
  for (int i = 0; i < 8; i++) {
    int r = rbase + i;
    if (r < Mrows) H[(long)r * Nc + colB] = acc[i] + bv;
  }
}

__global__ void k_zero(float* __restrict__ p, int n) {
  int t = blockIdx.x * blockDim.x + threadIdx.x;
  if (t < n) p[t] = 0.f;
}

// Coalesced partial sums: thread (g, c) with c fastest; atomics into sums[0..2C).
__global__ void k_bn_partial(const float* __restrict__ H, float* __restrict__ sums,
                             long Mrows, int C, int G) {
  int tid = blockIdx.x * blockDim.x + threadIdx.x;
  if (tid >= C * G) return;
  int c = tid % C;
  int g = tid / C;
  float s = 0.f, s2 = 0.f;
  for (long r = g; r < Mrows; r += G) {
    float v = H[r * C + c];
    s += v; s2 += v * v;
  }
  atomicAdd(&sums[c], s);
  atomicAdd(&sums[C + c], s2);
}

__global__ void k_bn_final(const float* __restrict__ sums, float* __restrict__ mean,
                           float* __restrict__ var, long Mrows, int C) {
  int c = blockIdx.x * blockDim.x + threadIdx.x;
  if (c >= C) return;
  float m = sums[c] / (float)Mrows;
  mean[c] = m;
  var[c] = sums[C + c] / (float)Mrows - m * m;
}

// BN + ReLU, f32 -> padded f16 activation for next GEMM / maxpool.
__global__ void k_bn_relu_f16(const float* __restrict__ H, _Float16* __restrict__ out,
                              const float* __restrict__ mean, const float* __restrict__ var,
                              const float* __restrict__ gamma, const float* __restrict__ beta,
                              long rows, int C, long rowsPad) {
  long tid = (long)blockIdx.x * blockDim.x + threadIdx.x;
  long total = rowsPad * C;
  if (tid >= total) return;
  int  c = (int)(tid % C);
  long r = tid / C;
  float v = 0.f;
  if (r < rows) {
    float h = H[r * C + c];
    v = fmaxf(gamma[c] * (h - mean[c]) * rsqrtf(var[c] + 1e-5f) + beta[c], 0.f);
  }
  out[tid] = (_Float16)v;
}

// Max over K from f16 activation; write f32 into channel slice of concat buffer.
__global__ void k_maxpool_f16(const _Float16* __restrict__ G, float* __restrict__ out,
                              int BM, int K, int C, int outStride, int outOff) {
  int tid = blockIdx.x * blockDim.x + threadIdx.x;
  if (tid >= BM * C) return;
  int c = tid % C;
  int bm = tid / C;
  float m = -FLT_MAX;
  const _Float16* g = &G[(long)bm * K * C + c];
  for (int k = 0; k < K; k++) m = fmaxf(m, (float)g[(long)k * C]);
  out[(long)bm * outStride + outOff + c] = m;
}

// Tiny final FC (256 -> 7) from f16 activation.
__global__ void k_fc_small(const _Float16* __restrict__ X, const float* __restrict__ W,
                           const float* __restrict__ b, float* __restrict__ out,
                           int M, int Kd, int N, int lda) {
  int tid = blockIdx.x * blockDim.x + threadIdx.x;
  if (tid >= M * N) return;
  int m = tid / N, n = tid % N;
  float s = b[n];
  for (int k = 0; k < Kd; k++) s += (float)X[(long)m * lda + k] * W[(long)k * N + n];
  out[tid] = s;
}

// ---------------------------------------------------------------------------

extern "C" void kernel_launch(void* const* d_in, const int* in_sizes, int n_in,
                              void* d_out, int out_size, void* d_ws, size_t ws_size,
                              hipStream_t stream) {
  (void)in_sizes; (void)n_in; (void)out_size; (void)ws_size;
  const int B = 8, N = 2048;
  auto r16 = [](long x) { return (x + 15) & ~15L; };
  auto r32 = [](int x)  { return (x + 31) & ~31; };

  // ---- parameter tensor mapping (setup_inputs dict insertion order) ----
  // d_in[0]=xyz(8,3,2048) d_in[1]=xy(8,2,2048); params: sa1,sa2,sa3,sa1x,sa2x,
  // sa3x,fc1,fc2,fc3; each layer W,b,gamma,beta (fc3: W,b). 180 tensors total.
  struct LinP { const float *W, *b, *g, *bt; int din, dout; };
  int cur = 2;
  auto take = [&](int din, int dout, bool bn) -> LinP {
    LinP L;
    L.W = (const float*)d_in[cur++];
    L.b = (const float*)d_in[cur++];
    if (bn) { L.g = (const float*)d_in[cur++]; L.bt = (const float*)d_in[cur++]; }
    else    { L.g = nullptr; L.bt = nullptr; }
    L.din = din; L.dout = dout;
    return L;
  };
  static const int M1[3][3] = {{32,32,64},{64,64,128},{64,96,128}};
  static const int M2[3][3] = {{64,64,128},{128,128,256},{128,128,256}};
  LinP sa1[3][3], sa2[3][3], sa3[3], sa1x[3][3], sa2x[3][3], sa3x[3], fc1, fc2, fc3;
  for (int i = 0; i < 3; i++) { int d = 3;   for (int l = 0; l < 3; l++) { sa1[i][l]  = take(d, M1[i][l], true); d = M1[i][l]; } }
  for (int i = 0; i < 3; i++) { int d = 323; for (int l = 0; l < 3; l++) { sa2[i][l]  = take(d, M2[i][l], true); d = M2[i][l]; } }
  { int d = 643; int w[3] = {256,256,512}; for (int l = 0; l < 3; l++) { sa3[l]  = take(d, w[l], true); d = w[l]; } }
  for (int i = 0; i < 3; i++) { int d = 2;   for (int l = 0; l < 3; l++) { sa1x[i][l] = take(d, M1[i][l], true); d = M1[i][l]; } }
  for (int i = 0; i < 3; i++) { int d = 322; for (int l = 0; l < 3; l++) { sa2x[i][l] = take(d, M2[i][l], true); d = M2[i][l]; } }
  { int d = 642; int w[3] = {256,256,512}; for (int l = 0; l < 3; l++) { sa3x[l] = take(d, w[l], true); d = w[l]; } }
  fc1 = take(1024, 512, true);
  fc2 = take(512, 256, true);
  fc3 = take(256, 7, false);

  // ---- workspace bump allocator ----
  size_t off = 0;
  auto allocB = [&](size_t bytes) -> void* {
    void* p = (char*)d_ws + off;
    off += (bytes + 255) & ~(size_t)255;
    return p;
  };
  auto allocF = [&](size_t n) { return (float*)allocB(n * sizeof(float)); };
  auto allocH = [&](size_t n) { return (_Float16*)allocB(n * sizeof(_Float16)); };
  auto allocI = [&](size_t n) { return (int*)allocB(n * sizeof(int)); };

  const size_t BIG = (size_t)B * 512 * 128 * 128;     // 67.1M elements
  float*    p3   = allocF((size_t)B * N * 3);
  float*    p2   = allocF((size_t)B * N * 2);
  int*      fpsI = allocI((size_t)B * 512);
  int*      gi   = allocI((size_t)B * 512 * 128);
  float*    l1x  = allocF((size_t)B * 512 * 3);
  float*    l1p  = allocF((size_t)B * 512 * 320);
  float*    l2x  = allocF((size_t)B * 128 * 3);
  float*    l2p  = allocF((size_t)B * 128 * 640);
  float*    m1x  = allocF((size_t)B * 512 * 2);
  float*    m1p  = allocF((size_t)B * 512 * 320);
  float*    m2x  = allocF((size_t)B * 128 * 2);
  float*    m2p  = allocF((size_t)B * 128 * 640);
  float*    mean = allocF(512);
  float*    var  = allocF(512);
  float*    sums = allocF(1024);
  float*    hx   = allocF((size_t)B * 1024);
  _Float16* Wp   = allocH((size_t)1024 * 512);        // largest packed weight (fc1)
  _Float16* AhA  = allocH(BIG);                       // activation ping (f16, padded)
  _Float16* AhB  = allocH(BIG);                       // activation pong
  float*    Hf   = allocF(BIG);                       // GEMM f32 output

  auto grid1 = [&](size_t n) { return dim3((unsigned)((n + TPB - 1) / TPB)); };

  auto gemm = [&](const _Float16* A, const LinP& L, int cinReal, int Kpad,
                  long rows, float* H) {
    int Nc = L.dout;
    long wtot = (long)(Nc / 16) * (Kpad / 32) * 512;
    k_pack_w<<<grid1((size_t)wtot), dim3(TPB), 0, stream>>>(L.W, Wp, cinReal, Nc, Kpad);
    int mt = (int)((rows + 15) / 16), nt = Nc / 16;
    int tiles = mt * nt;
    int blocks = (tiles + 7) / 8;   // 8 waves per 256-thread block
    k_gemm_wmma<<<dim3(blocks), dim3(256), 0, stream>>>(A, Wp, L.b, H, (int)rows, Kpad, Nc, tiles);
  };
  auto bn = [&](const float* H, _Float16* outA, const LinP& L, long rows, long rowsPad) {
    int C = L.dout;
    k_zero<<<grid1((size_t)2 * C), dim3(TPB), 0, stream>>>(sums, 2 * C);
    int G = (int)((262144 / C) < rows ? (262144 / C) : rows);
    if (G < 1) G = 1;
    k_bn_partial<<<grid1((size_t)C * G), dim3(TPB), 0, stream>>>(H, sums, rows, C, G);
    k_bn_final<<<grid1((size_t)C), dim3(TPB), 0, stream>>>(sums, mean, var, rows, C);
    k_bn_relu_f16<<<grid1((size_t)rowsPad * C), dim3(TPB), 0, stream>>>(H, outA, mean, var, L.g, L.bt, rows, C, rowsPad);
  };

  // ---- set abstraction with multi-scale grouping ----
  auto run_sa = [&](const float* pts, int Nin, int Cx, const float* feats, int Cf,
                    int npoint, const float* Rr, const int* Kk, LinP (*br)[3],
                    float* nx, float* catout, int totC) {
    k_fps<<<dim3(B), dim3(TPB), 0, stream>>>(pts, fpsI, Nin, Cx, npoint);
    k_gather<<<grid1((size_t)B * npoint * Cx), dim3(TPB), 0, stream>>>(pts, fpsI, nx, B, Nin, Cx, npoint);
    int coff = 0;
    for (int i = 0; i < 3; i++) {
      int K = Kk[i];
      float r2v = Rr[i] * Rr[i];
      k_query_ball<<<grid1((size_t)B * npoint), dim3(TPB), 0, stream>>>(pts, nx, gi, B, Nin, Cx, npoint, K, r2v);
      long rows = (long)B * npoint * K;
      long rowsPad = r16(rows);
      int C0 = Cf + Cx, C0p = r32(C0);
      k_group_f16<<<grid1((size_t)rowsPad * C0p), dim3(TPB), 0, stream>>>(pts, feats, nx, gi, AhA, B, Nin, Cx, Cf, npoint, K, C0p, rowsPad);
      _Float16* curp = AhA; _Float16* nxt = AhB;
      int cin = C0, cinPad = C0p;
      for (int l = 0; l < 3; l++) {
        const LinP& L = br[i][l];
        gemm(curp, L, cin, cinPad, rows, Hf);
        bn(Hf, nxt, L, rows, rowsPad);
        _Float16* t = curp; curp = nxt; nxt = t;
        cin = L.dout; cinPad = L.dout;   // widths are all multiples of 32
      }
      k_maxpool_f16<<<grid1((size_t)B * npoint * cin), dim3(TPB), 0, stream>>>(curp, catout, B * npoint, K, cin, totC, coff);
      coff += cin;
    }
  };

  // ---- group-all set abstraction ----
  auto run_all = [&](const float* xyzB, int M, int Cx, const float* featsB, int Cf,
                     LinP* lay, float* dst, int dstStride, int dstOff) {
    long rows = (long)B * M;
    long rowsPad = r16(rows);
    int C0 = Cx + Cf, C0p = r32(C0);
    k_concat2_f16<<<grid1((size_t)rowsPad * C0p), dim3(TPB), 0, stream>>>(xyzB, featsB, AhA, rows, Cx, Cf, C0p, rowsPad);
    _Float16* curp = AhA; _Float16* nxt = AhB;
    int cin = C0, cinPad = C0p;
    for (int l = 0; l < 3; l++) {
      const LinP& L = lay[l];
      gemm(curp, L, cin, cinPad, rows, Hf);
      bn(Hf, nxt, L, rows, rowsPad);
      _Float16* t = curp; curp = nxt; nxt = t;
      cin = L.dout; cinPad = L.dout;
    }
    k_maxpool_f16<<<grid1((size_t)B * cin), dim3(TPB), 0, stream>>>(curp, dst, B, M, cin, dstStride, dstOff);
  };

  // ---- pipeline ----
  const float* xyz = (const float*)d_in[0];
  const float* xy  = (const float*)d_in[1];
  k_transpose<<<grid1((size_t)B * 3 * N), dim3(TPB), 0, stream>>>(xyz, p3, B, 3, N);
  k_transpose<<<grid1((size_t)B * 2 * N), dim3(TPB), 0, stream>>>(xy,  p2, B, 2, N);

  static const float R1[3] = {0.1f, 0.2f, 0.4f}; static const int K1[3] = {16, 32, 128};
  static const float R2[3] = {0.2f, 0.4f, 0.8f}; static const int K2[3] = {32, 64, 128};

  run_sa(p3,  N,   3, nullptr, 0,   512, R1, K1, sa1,  l1x, l1p, 320);
  run_sa(l1x, 512, 3, l1p,     320, 128, R2, K2, sa2,  l2x, l2p, 640);
  run_all(l2x, 128, 3, l2p, 640, sa3, hx, 1024, 0);      // f3 -> hx[:, 0:512]

  run_sa(p2,  N,   2, nullptr, 0,   512, R1, K1, sa1x, m1x, m1p, 320);
  run_sa(m1x, 512, 2, m1p,     320, 128, R2, K2, sa2x, m2x, m2p, 640);
  run_all(m2x, 128, 2, m2p, 640, sa3x, hx, 1024, 512);   // f2 -> hx[:, 512:1024]

  // ---- head (rows=8, padded to 16) ----
  {
    long rows = B, rowsPad = 16;
    k_concat2_f16<<<grid1((size_t)rowsPad * 1024), dim3(TPB), 0, stream>>>(hx, nullptr, AhA, rows, 1024, 0, 1024, rowsPad);
    gemm(AhA, fc1, 1024, 1024, rows, Hf);
    bn(Hf, AhB, fc1, rows, rowsPad);
    gemm(AhB, fc2, 512, 512, rows, Hf);
    bn(Hf, AhA, fc2, rows, rowsPad);
    k_fc_small<<<grid1((size_t)B * 7), dim3(TPB), 0, stream>>>(AhA, fc3.W, fc3.b, (float*)d_out, B, 256, 7, 256);
  }
}
